// DecoderTransformer_41558103556621
// MI455X (gfx1250) — matrix operand order repaired and checked by reading, not compiled, not run on hardware
//
#include <hip/hip_runtime.h>
#include <hip/hip_bf16.h>

// ---------------- model dims ----------------
static constexpr int BB = 2, TT = 1024, EE = 768, HH = 12, LL = 12;
static constexpr int VV = 50257, FF = 3072, DD = 64;
static constexpr int MM = BB * TT; // 2048 rows

// ---------------- types ----------------
typedef __bf16  bf16_t;
typedef bf16_t  v16bf __attribute__((ext_vector_type(16)));
typedef float   v8f   __attribute__((ext_vector_type(8)));
typedef float   v4f   __attribute__((ext_vector_type(4)));
typedef unsigned int v4u __attribute__((ext_vector_type(4)));
typedef unsigned int v8u __attribute__((ext_vector_type(8)));

__device__ __forceinline__ unsigned short f32_to_bf16(float f) {
    unsigned int u = __float_as_uint(f);
    unsigned int r = (u + 0x7FFFu + ((u >> 16) & 1u)) >> 16;   // RNE
    return (unsigned short)r;
}

__device__ __forceinline__ int imin(int a, int b) { return a < b ? a : b; }

// Build a 16-element bf16 fragment from two 16B loads: p[0..7] and (p+delta)[0..7].
// A-fragment: delta = 8  (contiguous: element e -> K = k0+hi*16+e, base offset hi*16)
// B-fragment: delta = 8  (contiguous, same as A: element e -> K = k0+khi+e)
__device__ __forceinline__ v16bf frag16(const unsigned short* __restrict__ p, int delta) {
    v4u lo = *(const v4u*)(p);
    v4u hw = *(const v4u*)(p + delta);
    v8u w;
#pragma unroll
    for (int j = 0; j < 4; ++j) { w[j] = lo[j]; w[4 + j] = hw[j]; }
    return __builtin_bit_cast(v16bf, w);
}

// ---------------- embedding ----------------
__global__ void gpt_embed_kernel(const int* __restrict__ idx, const float* __restrict__ wte,
                                 const float* __restrict__ wpe, float* __restrict__ x, int total) {
    int i = blockIdx.x * blockDim.x + threadIdx.x;
    if (i >= total) return;
    int e  = i % EE;
    int bt = i / EE;
    int t  = bt % TT;
    x[i] = wte[(size_t)idx[bt] * EE + e] + wpe[(size_t)t * EE + e];
}

// ---------------- layernorm -> bf16 activations ----------------
__global__ void gpt_layernorm_kernel(const float* __restrict__ x, const float* __restrict__ g,
                                     const float* __restrict__ b, unsigned short* __restrict__ out) {
    int row = blockIdx.x;
    const float* xr = x + (size_t)row * EE;
    __shared__ float red[256];
    int tid = threadIdx.x;
    float s = 0.f;
    for (int i = tid; i < EE; i += 256) s += xr[i];
    red[tid] = s; __syncthreads();
    for (int o = 128; o > 0; o >>= 1) { if (tid < o) red[tid] += red[tid + o]; __syncthreads(); }
    float mu = red[0] / EE; __syncthreads();
    float v = 0.f;
    for (int i = tid; i < EE; i += 256) { float d = xr[i] - mu; v += d * d; }
    red[tid] = v; __syncthreads();
    for (int o = 128; o > 0; o >>= 1) { if (tid < o) red[tid] += red[tid + o]; __syncthreads(); }
    float rstd = rsqrtf(red[0] / EE + 1e-5f); __syncthreads();
    for (int i = tid; i < EE; i += 256)
        out[(size_t)row * EE + i] = f32_to_bf16((xr[i] - mu) * rstd * g[i] + b[i]);
}

// ---------------- fp32 -> bf16 (elementwise) ----------------
__global__ void gpt_convert_kernel(const float* __restrict__ in, unsigned short* __restrict__ out, size_t n) {
    size_t i = (size_t)blockIdx.x * blockDim.x + threadIdx.x;
    if (i < n) out[i] = f32_to_bf16(in[i]);
}

// ---------------- weight K x N fp32 -> N x K bf16 (transpose + convert) ----------------
__global__ void gpt_convert_wT_kernel(const float* __restrict__ w, unsigned short* __restrict__ wT,
                                      int K, int N) {
    size_t i = (size_t)blockIdx.x * blockDim.x + threadIdx.x;
    if (i >= (size_t)K * N) return;
    int n = (int)(i / K);
    int k = (int)(i % K);
    wT[i] = f32_to_bf16(w[(size_t)k * N + n]);
}

// ---------------- WMMA GEMM: C[MxN] (+=) A[MxK]bf16 * B[NxK]bf16^T + bias ----------------
// one wave per 16x64 output strip; A fragment reused across 4 N-tiles.
// Row pointers hoisted out of the K loop (clamped n is loop-invariant; OOB
// columns only pollute D elements that are never stored).
__global__ void gpt_gemm_bf16_kernel(const unsigned short* __restrict__ A,
                                     const unsigned short* __restrict__ Bm,
                                     const float* __restrict__ bias,
                                     float* __restrict__ C,
                                     int M, int N, int K, int accum) {
    int n0   = blockIdx.x * 64;
    int m0   = blockIdx.y * 16;
    int lane = threadIdx.x;
    int nn   = lane & 15;
    int hi   = (lane >> 4) & 1;
    int khi  = hi * 16;

    const unsigned short* arow = A + (size_t)(m0 + nn) * K + hi * 16;
    const unsigned short* br0 = Bm + (size_t)imin(n0 + nn,      N - 1) * K + khi;
    const unsigned short* br1 = Bm + (size_t)imin(n0 + 16 + nn, N - 1) * K + khi;
    const unsigned short* br2 = Bm + (size_t)imin(n0 + 32 + nn, N - 1) * K + khi;
    const unsigned short* br3 = Bm + (size_t)imin(n0 + 48 + nn, N - 1) * K + khi;

    v8f acc0, acc1, acc2, acc3;
#pragma unroll
    for (int r = 0; r < 8; ++r) { acc0[r] = 0.f; acc1[r] = 0.f; acc2[r] = 0.f; acc3[r] = 0.f; }

    for (int k0 = 0; k0 < K; k0 += 32) {
        v16bf a  = frag16(arow + k0, 8);
        v16bf b0 = frag16(br0 + k0, 8);
        v16bf b1 = frag16(br1 + k0, 8);
        v16bf b2 = frag16(br2 + k0, 8);
        v16bf b3 = frag16(br3 + k0, 8);
        acc0 = __builtin_amdgcn_wmma_f32_16x16x32_bf16(false, a, false, b0, (short)0, acc0, false, false);
        acc1 = __builtin_amdgcn_wmma_f32_16x16x32_bf16(false, a, false, b1, (short)0, acc1, false, false);
        acc2 = __builtin_amdgcn_wmma_f32_16x16x32_bf16(false, a, false, b2, (short)0, acc2, false, false);
        acc3 = __builtin_amdgcn_wmma_f32_16x16x32_bf16(false, a, false, b3, (short)0, acc3, false, false);
    }
#pragma unroll
    for (int t = 0; t < 4; ++t) {
        v8f acc = (t == 0) ? acc0 : (t == 1) ? acc1 : (t == 2) ? acc2 : acc3;
        int n = n0 + 16 * t + nn;
        if (n < N) {
            float bv = bias ? bias[n] : 0.f;
#pragma unroll
            for (int r = 0; r < 8; ++r) {
                int m = m0 + hi * 8 + r;
                size_t idx = (size_t)m * N + n;
                float v = acc[r] + bv;
                if (accum) v += C[idx];
                C[idx] = v;
            }
        }
    }
}

// ---------------- qkv split: [M,3E] fp32 -> q,k [B,H,T,D] bf16, v^T [B,H,D,T] bf16 ----------------
__global__ void gpt_qkv_split_kernel(const float* __restrict__ qkv, unsigned short* __restrict__ qb,
                                     unsigned short* __restrict__ kb, unsigned short* __restrict__ vtb) {
    int i = blockIdx.x * blockDim.x + threadIdx.x;
    if (i >= MM * EE) return;
    int d    = i % DD;
    int rest = i / DD;
    int h    = rest % HH;
    int bt   = rest / HH;
    int t    = bt % TT;
    int b    = bt / TT;
    size_t base = (size_t)bt * (3 * EE);
    float q = qkv[base + h * DD + d];
    float k = qkv[base + EE + h * DD + d];
    float v = qkv[base + 2 * EE + h * DD + d];
    size_t bh = (size_t)(b * HH + h);
    qb[(bh * TT + t) * DD + d]  = f32_to_bf16(q);
    kb[(bh * TT + t) * DD + d]  = f32_to_bf16(k);
    vtb[(bh * DD + d) * TT + t] = f32_to_bf16(v);
}

// ---------------- flash attention (WMMA), one wave per (b,h, 16-query tile) ----------------
// Key-block loop: largest kblk (multiple of 32) <= qt0+15 <= T-1, so all key
// reads stay < T. No clamping needed; causal mask zeroes invalid columns.
__global__ void gpt_attn_kernel(const unsigned short* __restrict__ qb,
                                const unsigned short* __restrict__ kb,
                                const unsigned short* __restrict__ vtb,
                                unsigned short* __restrict__ yb) {
    int ntiles = TT / 16;
    int qtile  = blockIdx.x % ntiles;
    int bh     = blockIdx.x / ntiles;
    int b      = bh / HH;
    int h      = bh % HH;
    int lane   = threadIdx.x;
    int qt0    = qtile * 16;
    int nn     = lane & 15;
    int hi     = (lane >> 4) & 1;
    int khi    = hi * 16;

    const unsigned short* Q  = qb  + (size_t)bh * TT * DD;
    const unsigned short* Kp = kb  + (size_t)bh * TT * DD;
    const unsigned short* Vt = vtb + (size_t)bh * DD * TT;

    v16bf aq0 = frag16(Q + (size_t)(qt0 + nn) * DD + hi * 16, 8);
    v16bf aq1 = frag16(Q + (size_t)(qt0 + nn) * DD + 32 + hi * 16, 8);

    // running pointers, stepped by one 32-key block per iteration
    const unsigned short* pk0 = Kp + (size_t)nn * DD + khi;        // keys kblk+nn
    const unsigned short* pk1 = Kp + (size_t)(16 + nn) * DD + khi; // keys kblk+16+nn
    const unsigned short* pv0 = Vt + (size_t)(nn)      * TT + khi; // d = nn
    const unsigned short* pv1 = Vt + (size_t)(16 + nn) * TT + khi;
    const unsigned short* pv2 = Vt + (size_t)(32 + nn) * TT + khi;
    const unsigned short* pv3 = Vt + (size_t)(48 + nn) * TT + khi;

    float rowmax[8], rowsum[8];
    v8f acc0, acc1, acc2, acc3;
#pragma unroll
    for (int r = 0; r < 8; ++r) {
        rowmax[r] = -1e30f; rowsum[r] = 0.f;
        acc0[r] = 0.f; acc1[r] = 0.f; acc2[r] = 0.f; acc3[r] = 0.f;
    }

    const float scale = 0.125f; // 1/sqrt(64)
    __shared__ float pshm[16 * 32];

    for (int kblk = 0; kblk <= qt0 + 15; kblk += 32) {
        v8f s0, s1;
#pragma unroll
        for (int r = 0; r < 8; ++r) { s0[r] = 0.f; s1[r] = 0.f; }
        {
            v16bf bk0 = frag16(pk0,      8);   // d chunk 0..31
            v16bf bk1 = frag16(pk0 + 32, 8);   // d chunk 32..63
            v16bf bk2 = frag16(pk1,      8);
            v16bf bk3 = frag16(pk1 + 32, 8);
            s0 = __builtin_amdgcn_wmma_f32_16x16x32_bf16(false, aq0, false, bk0, (short)0, s0, false, false);
            s0 = __builtin_amdgcn_wmma_f32_16x16x32_bf16(false, aq1, false, bk1, (short)0, s0, false, false);
            s1 = __builtin_amdgcn_wmma_f32_16x16x32_bf16(false, aq0, false, bk2, (short)0, s1, false, false);
            s1 = __builtin_amdgcn_wmma_f32_16x16x32_bf16(false, aq1, false, bk3, (short)0, s1, false, false);
        }
        // online softmax over this 16x32 score block
#pragma unroll
        for (int r = 0; r < 8; ++r) {
            int m    = hi * 8 + r;
            int qrow = qt0 + m;
            int key0 = kblk + nn;
            int key1 = kblk + 16 + nn;
            float v0 = (key0 <= qrow) ? s0[r] * scale : -1e30f;
            float v1 = (key1 <= qrow) ? s1[r] * scale : -1e30f;
            float bm = fmaxf(v0, v1);
            for (int o = 8; o >= 1; o >>= 1) bm = fmaxf(bm, __shfl_xor(bm, o, 32));
            float nm   = fmaxf(rowmax[r], bm);
            float corr = __expf(rowmax[r] - nm);
            float p0   = __expf(v0 - nm);
            float p1   = __expf(v1 - nm);
            float ps   = p0 + p1;
            for (int o = 8; o >= 1; o >>= 1) ps += __shfl_xor(ps, o, 32);
            rowsum[r] = rowsum[r] * corr + ps;
            rowmax[r] = nm;
            acc0[r] *= corr; acc1[r] *= corr; acc2[r] *= corr; acc3[r] *= corr;
            pshm[m * 32 + nn]      = p0;
            pshm[m * 32 + 16 + nn] = p1;
        }
        __syncthreads();
        // P (16x32) -> bf16 A fragment (rows of pshm are 32B-aligned)
        v16bf ap;
        {
            const v4f* pr0 = (const v4f*)&pshm[nn * 32 + hi * 16];
            const v4f* pr1 = (const v4f*)&pshm[nn * 32 + hi * 16 + 8];
            v4f f0 = pr0[0], f1 = pr0[1], f2 = pr1[0], f3 = pr1[1];
#pragma unroll
            for (int j = 0; j < 4; ++j) {
                ap[j]      = __builtin_bit_cast(bf16_t, f32_to_bf16(f0[j]));
                ap[4 + j]  = __builtin_bit_cast(bf16_t, f32_to_bf16(f1[j]));
                ap[8 + j]  = __builtin_bit_cast(bf16_t, f32_to_bf16(f2[j]));
                ap[12 + j] = __builtin_bit_cast(bf16_t, f32_to_bf16(f3[j]));
            }
        }
        __syncthreads();
        // acc += P @ V  (V block is 32 x 64)
        {
            v16bf bv0 = frag16(pv0, 8);
            v16bf bv1 = frag16(pv1, 8);
            v16bf bv2 = frag16(pv2, 8);
            v16bf bv3 = frag16(pv3, 8);
            acc0 = __builtin_amdgcn_wmma_f32_16x16x32_bf16(false, ap, false, bv0, (short)0, acc0, false, false);
            acc1 = __builtin_amdgcn_wmma_f32_16x16x32_bf16(false, ap, false, bv1, (short)0, acc1, false, false);
            acc2 = __builtin_amdgcn_wmma_f32_16x16x32_bf16(false, ap, false, bv2, (short)0, acc2, false, false);
            acc3 = __builtin_amdgcn_wmma_f32_16x16x32_bf16(false, ap, false, bv3, (short)0, acc3, false, false);
        }
        pk0 += 32 * DD; pk1 += 32 * DD;
        pv0 += 32; pv1 += 32; pv2 += 32; pv3 += 32;
    }
    // normalize + write y as bf16 in [B,T,E]
#pragma unroll
    for (int c = 0; c < 4; ++c) {
        v8f acc = (c == 0) ? acc0 : (c == 1) ? acc1 : (c == 2) ? acc2 : acc3;
#pragma unroll
        for (int r = 0; r < 8; ++r) {
            int m = hi * 8 + r;
            float inv = 1.f / rowsum[r];
            size_t o = ((size_t)(b * TT + qt0 + m)) * EE + h * DD + c * 16 + nn;
            yb[o] = f32_to_bf16(acc[r] * inv);
        }
    }
}

// ---------------- gelu(tanh) fp32 -> bf16 ----------------
__global__ void gpt_gelu_kernel(const float* __restrict__ in, unsigned short* __restrict__ out, size_t n) {
    size_t i = (size_t)blockIdx.x * blockDim.x + threadIdx.x;
    if (i >= n) return;
    float x = in[i];
    float c = 0.7978845608028654f * (x + 0.044715f * x * x * x);
    out[i]  = f32_to_bf16(0.5f * x * (1.f + tanhf(c)));
}

// ---------------- cross-entropy loss ----------------
__global__ void gpt_loss_kernel(const float* __restrict__ logits, const int* __restrict__ targets,
                                float* __restrict__ loss, float inv_rows) {
    int row = blockIdx.x;
    const float* lr = logits + (size_t)row * VV;
    __shared__ float red[256];
    int tid = threadIdx.x;
    float mx = -1e30f;
    for (int i = tid; i < VV; i += 256) mx = fmaxf(mx, lr[i]);
    red[tid] = mx; __syncthreads();
    for (int o = 128; o > 0; o >>= 1) { if (tid < o) red[tid] = fmaxf(red[tid], red[tid + o]); __syncthreads(); }
    mx = red[0]; __syncthreads();
    float s = 0.f;
    for (int i = tid; i < VV; i += 256) s += __expf(lr[i] - mx);
    red[tid] = s; __syncthreads();
    for (int o = 128; o > 0; o >>= 1) { if (tid < o) red[tid] += red[tid + o]; __syncthreads(); }
    if (tid == 0) {
        float lse = __logf(red[0]) + mx;
        float tgt = lr[targets[row]];
        atomicAdd(loss, (lse - tgt) * inv_rows);
    }
}

// ---------------- launch ----------------
extern "C" void kernel_launch(void* const* d_in, const int* in_sizes, int n_in,
                              void* d_out, int out_size, void* d_ws, size_t ws_size,
                              hipStream_t stream) {
    const int*   idx       = (const int*)d_in[0];
    const int*   targets   = (const int*)d_in[1];
    const float* wte       = (const float*)d_in[2];
    const float* wpe       = (const float*)d_in[3];
    const float* ln1_g     = (const float*)d_in[4];
    const float* ln1_b     = (const float*)d_in[5];
    const float* w_qkv     = (const float*)d_in[6];
    const float* b_qkv     = (const float*)d_in[7];
    const float* w_attproj = (const float*)d_in[8];
    const float* b_attproj = (const float*)d_in[9];
    const float* ln2_g     = (const float*)d_in[10];
    const float* ln2_b     = (const float*)d_in[11];
    const float* w_fc      = (const float*)d_in[12];
    const float* b_fc      = (const float*)d_in[13];
    const float* w_mlpproj = (const float*)d_in[14];
    const float* b_mlpproj = (const float*)d_in[15];

    float* logits = (float*)d_out;
    float* loss   = (float*)d_out + (size_t)MM * VV;

    char* ws = (char*)d_ws;
    auto carve = [&](size_t bytes) {
        void* p = (void*)ws;
        ws += (bytes + 255) & ~(size_t)255;
        return p;
    };
    float*          x    = (float*)carve((size_t)MM * EE * 4);
    unsigned short* act  = (unsigned short*)carve((size_t)MM * FF * 2);
    float*          qkvb = (float*)carve((size_t)MM * 3 * EE * 4);
    float*          fcb  = (float*)carve((size_t)MM * FF * 4);
    unsigned short* wT   = (unsigned short*)carve((size_t)EE * FF * 2);
    unsigned short* qb   = (unsigned short*)carve((size_t)MM * EE * 2);
    unsigned short* kbuf = (unsigned short*)carve((size_t)MM * EE * 2);
    unsigned short* vtb  = (unsigned short*)carve((size_t)MM * EE * 2);
    unsigned short* yb   = (unsigned short*)carve((size_t)MM * EE * 2);
    unsigned short* wteb = (unsigned short*)carve((size_t)VV * EE * 2);

    auto ceil_div = [](long long a, long long b) { return (int)((a + b - 1) / b); };

    // embedding
    gpt_embed_kernel<<<ceil_div((long long)MM * EE, 256), 256, 0, stream>>>(idx, wte, wpe, x, MM * EE);

    for (int l = 0; l < LL; ++l) {
        // --- attention block ---
        gpt_layernorm_kernel<<<MM, 256, 0, stream>>>(x, ln1_g + (size_t)l * EE, ln1_b + (size_t)l * EE, act);
        gpt_convert_wT_kernel<<<ceil_div((long long)EE * 3 * EE, 256), 256, 0, stream>>>(
            w_qkv + (size_t)l * EE * 3 * EE, wT, EE, 3 * EE);
        gpt_gemm_bf16_kernel<<<dim3(ceil_div(3 * EE, 64), MM / 16), 32, 0, stream>>>(
            act, wT, b_qkv + (size_t)l * 3 * EE, qkvb, MM, 3 * EE, EE, 0);
        gpt_qkv_split_kernel<<<ceil_div((long long)MM * EE, 256), 256, 0, stream>>>(qkvb, qb, kbuf, vtb);
        gpt_attn_kernel<<<BB * HH * (TT / 16), 32, 0, stream>>>(qb, kbuf, vtb, yb);
        gpt_convert_wT_kernel<<<ceil_div((long long)EE * EE, 256), 256, 0, stream>>>(
            w_attproj + (size_t)l * EE * EE, wT, EE, EE);
        gpt_gemm_bf16_kernel<<<dim3(ceil_div(EE, 64), MM / 16), 32, 0, stream>>>(
            yb, wT, b_attproj + (size_t)l * EE, x, MM, EE, EE, 1);

        // --- MLP block ---
        gpt_layernorm_kernel<<<MM, 256, 0, stream>>>(x, ln2_g + (size_t)l * EE, ln2_b + (size_t)l * EE, act);
        gpt_convert_wT_kernel<<<ceil_div((long long)EE * FF, 256), 256, 0, stream>>>(
            w_fc + (size_t)l * EE * FF, wT, EE, FF);
        gpt_gemm_bf16_kernel<<<dim3(ceil_div(FF, 64), MM / 16), 32, 0, stream>>>(
            act, wT, b_fc + (size_t)l * FF, fcb, MM, FF, EE, 0);
        gpt_gelu_kernel<<<ceil_div((long long)MM * FF, 256), 256, 0, stream>>>(fcb, act, (size_t)MM * FF);
        gpt_convert_wT_kernel<<<ceil_div((long long)FF * EE, 256), 256, 0, stream>>>(
            w_mlpproj + (size_t)l * FF * EE, wT, FF, EE);
        gpt_gemm_bf16_kernel<<<dim3(ceil_div(EE, 64), MM / 16), 32, 0, stream>>>(
            act, wT, b_mlpproj + (size_t)l * EE, x, MM, EE, FF, 1);
    }

    // LM head: logits = x @ wte^T   (wte is already [V,E] = N-major)
    gpt_convert_kernel<<<ceil_div((long long)MM * EE, 256), 256, 0, stream>>>(x, act, (size_t)MM * EE);
    gpt_convert_kernel<<<ceil_div((long long)VV * EE, 256), 256, 0, stream>>>(wte, wteb, (size_t)VV * EE);
    gpt_gemm_bf16_kernel<<<dim3(ceil_div(VV, 64), MM / 16), 32, 0, stream>>>(
        act, wteb, nullptr, logits, MM, VV, EE, 0);

    // loss
    hipMemsetAsync(loss, 0, sizeof(float), stream);
    gpt_loss_kernel<<<MM, 256, 0, stream>>>(logits, targets, loss, 1.f / (float)MM);
}
